// Residual_block_57140244906119
// MI455X (gfx1250) — matrix operand order repaired and checked by reading, not compiled, not run on hardware
//
#include <hip/hip_runtime.h>

// ---------------------------------------------------------------------------
// Residual block: relu(LN(x)@W1^T+b1) -> LN -> @W2^T+b2 -> +x -> relu
// bf16 WMMA (v_wmma_f32_16x16x32_bf16) GEMM core, fp32 accumulate.
// Global->LDS staging via GLOBAL_LOAD_ASYNC_TO_LDS_B128 (ASYNCcnt) when the
// toolchain exposes the builtin; 3-deep LDS ring -> one barrier per K-step.
// ---------------------------------------------------------------------------

#define HDIM 2048
#define MROWS 8192
#define BK 32
#define NT (HDIM / BK)          // 64 K-steps
#define LDSS 40                 // padded LDS row stride (80 B, conflict-free)
#define TILE (128 * LDSS)       // ushorts per tile buffer (10240 B)

typedef __bf16 bf16_t;
typedef bf16_t v16bf __attribute__((ext_vector_type(16)));
typedef float  v8f   __attribute__((ext_vector_type(8)));
typedef unsigned int u32x4 __attribute__((ext_vector_type(4)));
typedef int v4i __attribute__((ext_vector_type(4)));

#if defined(__HIP_DEVICE_COMPILE__) && __has_builtin(__builtin_amdgcn_global_load_async_to_lds_b128)
#define USE_ASYNC 1
#else
#define USE_ASYNC 0
#endif

#if USE_ASYNC
#define GPTR(p) ((__attribute__((address_space(1))) v4i*)(p))
#define LPTR(p) ((__attribute__((address_space(3))) v4i*)(p))
#if __has_builtin(__builtin_amdgcn_s_wait_asynccnt)
#define WAIT_ASYNC(n) __builtin_amdgcn_s_wait_asynccnt(n)
#else
#define WAIT_ASYNC(n) asm volatile("s_wait_asynccnt %0" ::"i"(n) : "memory")
#endif
#endif

__device__ __forceinline__ unsigned short f32_to_bf16(float f) {
    unsigned u = __builtin_bit_cast(unsigned, f);
    unsigned r = u + 0x7FFFu + ((u >> 16) & 1u);   // round to nearest even
    return (unsigned short)(r >> 16);
}
__device__ __forceinline__ float bf16_to_f32(unsigned short h) {
    unsigned u = ((unsigned)h) << 16;
    return __builtin_bit_cast(float, u);
}

// ---------------------------------------------------------------------------
// fp32 -> bf16 weight conversion (grid-stride)
// ---------------------------------------------------------------------------
__global__ __launch_bounds__(256)
void cvt_f32_bf16_kernel(const float* __restrict__ in,
                         unsigned short* __restrict__ out, int n) {
    int i = blockIdx.x * blockDim.x + threadIdx.x;
    int stride = gridDim.x * blockDim.x;
    for (; i < n; i += stride) out[i] = f32_to_bf16(in[i]);
}

// ---------------------------------------------------------------------------
// Row LayerNorm (H=2048) fused with bf16 cast. One block per row, 8 elems/thr.
// ---------------------------------------------------------------------------
__device__ __forceinline__ float ld_val(const float* p)          { return *p; }
__device__ __forceinline__ float ld_val(const unsigned short* p) { return bf16_to_f32(*p); }

template <typename TIN>
__global__ __launch_bounds__(256)
void ln_bf16_kernel(const TIN* __restrict__ in, const float* __restrict__ gamma,
                    const float* __restrict__ beta, unsigned short* __restrict__ out) {
    __shared__ float red[256];
    const int row = blockIdx.x;
    const int tid = threadIdx.x;
    const TIN* rp = in + (size_t)row * HDIM;

    float v[8];
    float s = 0.f, sq = 0.f;
#pragma unroll
    for (int i = 0; i < 8; ++i) {
        float f = ld_val(rp + tid + i * 256);
        v[i] = f; s += f; sq += f * f;
    }
    red[tid] = s; __syncthreads();
#pragma unroll
    for (int off = 128; off > 0; off >>= 1) {
        if (tid < off) red[tid] += red[tid + off];
        __syncthreads();
    }
    const float mean = red[0] * (1.0f / HDIM);
    __syncthreads();
    red[tid] = sq; __syncthreads();
#pragma unroll
    for (int off = 128; off > 0; off >>= 1) {
        if (tid < off) red[tid] += red[tid + off];
        __syncthreads();
    }
    const float var = red[0] * (1.0f / HDIM) - mean * mean;
    const float rs  = rsqrtf(var + 1e-5f);

    unsigned short* op = out + (size_t)row * HDIM;
#pragma unroll
    for (int i = 0; i < 8; ++i) {
        const int c = tid + i * 256;
        float f = (v[i] - mean) * rs * gamma[c] + beta[c];
        op[c] = f32_to_bf16(f);
    }
}

// ---------------------------------------------------------------------------
// Tiled WMMA GEMM:  C[M,N] = A[M,K](bf16) * W[N,K]^T(bf16) + bias
//   block tile 128x128, K-step 32, 8 waves (2 N-slices x 4 M-slices),
//   each wave -> 2x4 grid of 16x16x32 WMMA tiles.
//   RES=0: out = relu(C) as bf16      (GEMM1)
//   RES=1: out = relu(C + resid) f32  (GEMM2, writes d_out)
// ---------------------------------------------------------------------------
template <bool RES>
__global__ __launch_bounds__(256)
void gemm_wmma_kernel(const unsigned short* __restrict__ A,
                      const unsigned short* __restrict__ W,
                      const float* __restrict__ bias,
                      const float* __restrict__ resid,
                      unsigned short* __restrict__ outb,
                      float* __restrict__ outf) {
    __shared__ __attribute__((aligned(16))) unsigned short sA[3 * TILE];
    __shared__ __attribute__((aligned(16))) unsigned short sB[3 * TILE];

    const int tid  = threadIdx.x;
    const int lane = tid & 31;
    const int wave = tid >> 5;
    const int wm   = wave & 3;    // 32-row slice within block tile
    const int wn   = wave >> 2;   // 64-col slice within block tile
    const int half = lane >> 4;   // wave32 half
    const int l16  = lane & 15;

    const int bm = blockIdx.y * 128;
    const int bn = blockIdx.x * 128;

    // Global->LDS staging: each thread moves two 16-byte chunks per matrix.
    const int r0 = tid >> 2;           // rows r0 and r0+64
    const int g0 = (tid & 3) * 8;      // 8 bf16 per chunk
    const unsigned short* Ag = A + (size_t)(bm + r0) * HDIM + g0;
    const unsigned short* Wg = W + (size_t)(bn + r0) * HDIM + g0;
    const int so0 = r0 * LDSS + g0;
    const int so1 = (r0 + 64) * LDSS + g0;

    v8f acc[2][4];
#pragma unroll
    for (int i = 0; i < 2; ++i)
#pragma unroll
        for (int j = 0; j < 4; ++j)
#pragma unroll
            for (int v = 0; v < 8; ++v) acc[i][j][v] = 0.0f;

    const int ka0 = half ? 8 : 0;    // A frag: K {0..7,16..23} / {8..15,24..31}
    const int kb0 = half ? 8 : 0;    // B frag: K {0..7,16..23} / {8..15,24..31}

#if USE_ASYNC
    // -------- async-to-LDS path: 3-deep ring, one barrier per K-step -------
    {
        // prologue: group 0 into ring slot 0
        __builtin_amdgcn_global_load_async_to_lds_b128(GPTR(Ag), LPTR(&sA[so0]), 0, 0);
        __builtin_amdgcn_global_load_async_to_lds_b128(GPTR(Ag + (size_t)64 * HDIM), LPTR(&sA[so1]), 0, 0);
        __builtin_amdgcn_global_load_async_to_lds_b128(GPTR(Wg), LPTR(&sB[so0]), 0, 0);
        __builtin_amdgcn_global_load_async_to_lds_b128(GPTR(Wg + (size_t)64 * HDIM), LPTR(&sB[so1]), 0, 0);

        int cur = 0, nxt = 1;
        for (int kt = 0; kt < NT; ++kt) {
            // issue group kt+1 (unconditional; one-group overshoot stays inside d_ws
            // and lands in a ring slot that is never consumed)
            const unsigned short* Agn = Ag + (kt + 1) * BK;
            const unsigned short* Wgn = Wg + (kt + 1) * BK;
            unsigned short* dA = &sA[nxt * TILE];
            unsigned short* dB = &sB[nxt * TILE];
            __builtin_amdgcn_global_load_async_to_lds_b128(GPTR(Agn), LPTR(dA + so0), 0, 0);
            __builtin_amdgcn_global_load_async_to_lds_b128(GPTR(Agn + (size_t)64 * HDIM), LPTR(dA + so1), 0, 0);
            __builtin_amdgcn_global_load_async_to_lds_b128(GPTR(Wgn), LPTR(dB + so0), 0, 0);
            __builtin_amdgcn_global_load_async_to_lds_b128(GPTR(Wgn + (size_t)64 * HDIM), LPTR(dB + so1), 0, 0);

            WAIT_ASYNC(4);          // group kt complete; only group kt+1 in flight
            __syncthreads();        // all waves' group-kt data visible in LDS

            const unsigned short* pA = &sA[cur * TILE];
            const unsigned short* pB = &sB[cur * TILE];

            v16bf af[2];
#pragma unroll
            for (int i = 0; i < 2; ++i) {
                const unsigned short* p = pA + (wm * 32 + i * 16 + l16) * LDSS;
                union { v16bf v; u32x4 q[2]; } u;
                u.q[0] = *(const u32x4*)(p + ka0);
                u.q[1] = *(const u32x4*)(p + ka0 + 16);
                af[i] = u.v;
            }
            v16bf bfr[4];
#pragma unroll
            for (int j = 0; j < 4; ++j) {
                const unsigned short* p = pB + (wn * 64 + j * 16 + l16) * LDSS + kb0;
                union { v16bf v; u32x4 q[2]; } u;
                u.q[0] = *(const u32x4*)(p);
                u.q[1] = *(const u32x4*)(p + 16);
                bfr[j] = u.v;
            }
#pragma unroll
            for (int i = 0; i < 2; ++i)
#pragma unroll
                for (int j = 0; j < 4; ++j)
                    acc[i][j] = __builtin_amdgcn_wmma_f32_16x16x32_bf16(
                        false, af[i], false, bfr[j], (short)0, acc[i][j], false, false);

            cur = nxt;
            nxt = (nxt + 1 == 3) ? 0 : nxt + 1;
        }
    }
#else
    // -------- fallback: register-prefetch + single LDS buffer --------------
    {
        u32x4 pa0 = *(const u32x4*)(Ag);
        u32x4 pa1 = *(const u32x4*)(Ag + (size_t)64 * HDIM);
        u32x4 pb0 = *(const u32x4*)(Wg);
        u32x4 pb1 = *(const u32x4*)(Wg + (size_t)64 * HDIM);

        for (int kt = 0; kt < NT; ++kt) {
            __syncthreads();
            *(u32x4*)(&sA[so0]) = pa0;
            *(u32x4*)(&sA[so1]) = pa1;
            *(u32x4*)(&sB[so0]) = pb0;
            *(u32x4*)(&sB[so1]) = pb1;
            __syncthreads();

            if (kt + 1 < NT) {
                const unsigned short* Agn = Ag + (kt + 1) * BK;
                const unsigned short* Wgn = Wg + (kt + 1) * BK;
                pa0 = *(const u32x4*)(Agn);
                pa1 = *(const u32x4*)(Agn + (size_t)64 * HDIM);
                pb0 = *(const u32x4*)(Wgn);
                pb1 = *(const u32x4*)(Wgn + (size_t)64 * HDIM);
            }

            v16bf af[2];
#pragma unroll
            for (int i = 0; i < 2; ++i) {
                const unsigned short* p = &sA[(wm * 32 + i * 16 + l16) * LDSS];
                union { v16bf v; u32x4 q[2]; } u;
                u.q[0] = *(const u32x4*)(p + ka0);
                u.q[1] = *(const u32x4*)(p + ka0 + 16);
                af[i] = u.v;
            }
            v16bf bfr[4];
#pragma unroll
            for (int j = 0; j < 4; ++j) {
                const unsigned short* p = &sB[(wn * 64 + j * 16 + l16) * LDSS + kb0];
                union { v16bf v; u32x4 q[2]; } u;
                u.q[0] = *(const u32x4*)(p);
                u.q[1] = *(const u32x4*)(p + 16);
                bfr[j] = u.v;
            }
#pragma unroll
            for (int i = 0; i < 2; ++i)
#pragma unroll
                for (int j = 0; j < 4; ++j)
                    acc[i][j] = __builtin_amdgcn_wmma_f32_16x16x32_bf16(
                        false, af[i], false, bfr[j], (short)0, acc[i][j], false, false);
        }
    }
#endif

    // Epilogue. C-tile layout: lane half selects M {0..7}/{8..15}, l16 = N.
#pragma unroll
    for (int i = 0; i < 2; ++i) {
#pragma unroll
        for (int j = 0; j < 4; ++j) {
            const int col = bn + wn * 64 + j * 16 + l16;
            const float bv = bias[col];
#pragma unroll
            for (int v = 0; v < 8; ++v) {
                const int row = bm + wm * 32 + i * 16 + half * 8 + v;
                float r = acc[i][j][v] + bv;
                if constexpr (RES) {
                    r += resid[(size_t)row * HDIM + col];
                    r = r > 0.0f ? r : 0.0f;
                    outf[(size_t)row * HDIM + col] = r;
                } else {
                    r = r > 0.0f ? r : 0.0f;
                    outb[(size_t)row * HDIM + col] = f32_to_bf16(r);
                }
            }
        }
    }
}

// ---------------------------------------------------------------------------
// Launch: cvt(W1,W2) -> LN1+cast -> GEMM1(relu) -> LN2+cast -> GEMM2(+x,relu)
// ---------------------------------------------------------------------------
extern "C" void kernel_launch(void* const* d_in, const int* in_sizes, int n_in,
                              void* d_out, int out_size, void* d_ws, size_t ws_size,
                              hipStream_t stream) {
    const float* x   = (const float*)d_in[0];
    const float* W1  = (const float*)d_in[1];
    const float* b1  = (const float*)d_in[2];
    const float* W2  = (const float*)d_in[3];
    const float* b2  = (const float*)d_in[4];
    const float* g1  = (const float*)d_in[5];
    const float* be1 = (const float*)d_in[6];
    const float* g2  = (const float*)d_in[7];
    const float* be2 = (const float*)d_in[8];
    float* out = (float*)d_out;

    unsigned short* Wb1 = (unsigned short*)d_ws;                       //  8 MB
    unsigned short* Wb2 = Wb1 + (size_t)HDIM * HDIM;                   //  8 MB
    unsigned short* Abf = Wb2 + (size_t)HDIM * HDIM;                   // 32 MB
    unsigned short* Ybf = Abf + (size_t)MROWS * HDIM;                  // 32 MB

    dim3 blk(256);
    cvt_f32_bf16_kernel<<<dim3(2048), blk, 0, stream>>>(W1, Wb1, HDIM * HDIM);
    cvt_f32_bf16_kernel<<<dim3(2048), blk, 0, stream>>>(W2, Wb2, HDIM * HDIM);

    ln_bf16_kernel<float><<<dim3(MROWS), blk, 0, stream>>>(x, g1, be1, Abf);

    dim3 ggrid(HDIM / 128, MROWS / 128);  // (16, 64)
    gemm_wmma_kernel<false><<<ggrid, blk, 0, stream>>>(Abf, Wb1, b1, nullptr, Ybf, nullptr);

    ln_bf16_kernel<unsigned short><<<dim3(MROWS), blk, 0, stream>>>(Ybf, g2, be2, Abf);

    gemm_wmma_kernel<true><<<ggrid, blk, 0, stream>>>(Abf, Wb2, b2, x, nullptr, out);
}
